// STGCN_62371515072689
// MI455X (gfx1250) — compile-verified, matrix operand order checked by reading, not run on hardware
//
#include <hip/hip_runtime.h>
#include <hip/hip_bf16.h>
#include <math.h>

// ---------------- constants from the reference ----------------
#define Bq   512
#define Tq   12
#define Nq   200
#define Fq   16
#define GHq  64
#define OUTq 512
#define FCHq 512
#define NLq  3
#define XSTR (Tq*Nq + Fq)      // 2416 floats per x row
#define Mq   512               // LSTM rows after the gather shortcut (== B)
#define ZNq  (4*GHq)           // 256

typedef __attribute__((ext_vector_type(16))) __bf16 v16bf;
typedef __attribute__((ext_vector_type(8)))  float  v8f;

__device__ __forceinline__ v8f wmma_bf16(v16bf a, v16bf b, v8f c) {
    return __builtin_amdgcn_wmma_f32_16x16x32_bf16(false, a, false, b, (short)0, c, false, false);
}

// Load a 16-lane fragment from a row-major (16 x 64) bf16 tile, row stride 64.
// Contiguous k order, same for A and B: lanes 0-15 hold rows 0-15, K = kb*32 + 0..15;
// lanes 16-31 hold rows 0-15, K = kb*32 + 16..31 (16 contiguous bf16 per lane).
// Works for both global and LDS base pointers (address space inferred after inlining).
__device__ __forceinline__ v16bf ld_frag64(const __bf16* base, int lane, int kb) {
    const __bf16* p = base + (size_t)(lane & 15) * 64 + kb * 32 + ((lane >> 4) << 4);
    return *(const v16bf*)p;
}

__device__ __forceinline__ float sigmoidf_(float x) { return 1.0f / (1.0f + expf(-x)); }

// ---------------- 1) weight prep: transpose + fp32->bf16 ----------------
// WxT/WhT: (NL, 256, 64) = Wx[l]^T ; WoutT: (512, 64) = W_out^T
__global__ void k_prep(const float* __restrict__ Wx, const float* __restrict__ Wh,
                       const float* __restrict__ Wout,
                       __bf16* __restrict__ WxT, __bf16* __restrict__ WhT,
                       __bf16* __restrict__ WoutT) {
    int idx = blockIdx.x * blockDim.x + threadIdx.x;   // 0 .. 131071
    if (idx < 2 * NLq * ZNq * GHq) {                   // WxT + WhT region
        int half = NLq * ZNq * GHq;                    // 49152
        const float* src = (idx < half) ? Wx : Wh;
        __bf16* dst = (idx < half) ? WxT : WhT;
        int li = idx % half;
        int l  = li / (ZNq * GHq);
        int r  = li % (ZNq * GHq);
        int n  = r / GHq;
        int k  = r % GHq;
        dst[li] = (__bf16)src[(size_t)l * GHq * ZNq + (size_t)k * ZNq + n];
    } else {
        int li = idx - 2 * NLq * ZNq * GHq;            // 0 .. 32767
        int n = li / GHq;
        int k = li % GHq;
        WoutT[li] = (__bf16)Wout[(size_t)k * OUTq + n];
    }
}

// ---------------- 2) gathered graph conv ----------------
// xg_sel[t,b,h] = relu( y0*Wg[0,h] + y1*Wg[1,h] + b_g[h] ),
// y_k = dot(cheb[k, sid[b], :], x1[b,t,:])   -> seqA (T, 512, 64) bf16
__global__ void k_gconv(const float* __restrict__ x, const float* __restrict__ cheb,
                        const float* __restrict__ Wg, const float* __restrict__ bg,
                        __bf16* __restrict__ seqA) {
    int t = blockIdx.x, b = blockIdx.y, tid = threadIdx.x;   // blockDim = 64
    __shared__ float r0[64], r1[64];
    __shared__ int rsh;
    const float* xrow = x + (size_t)b * XSTR;
    if (tid == 0) rsh = (int)xrow[XSTR - 1];
    __syncthreads();
    int r = rsh;
    const float* c0 = cheb + (size_t)r * Nq;
    const float* c1 = cheb + (size_t)Nq * Nq + (size_t)r * Nq;
    const float* xv = xrow + t * Nq;
    float s0 = 0.f, s1 = 0.f;
    for (int m = tid; m < Nq; m += 64) {
        float xm = xv[m];
        s0 += c0[m] * xm;
        s1 += c1[m] * xm;
    }
    r0[tid] = s0; r1[tid] = s1;
    __syncthreads();
    for (int s = 32; s > 0; s >>= 1) {
        if (tid < s) { r0[tid] += r0[tid + s]; r1[tid] += r1[tid + s]; }
        __syncthreads();
    }
    float v = r0[0] * Wg[tid] + r1[0] * Wg[GHq + tid] + bg[tid];
    v = v > 0.f ? v : 0.f;
    seqA[((size_t)t * Mq + b) * GHq + tid] = (__bf16)v;
}

// ---------------- 3) persistent LSTM: all 3 layers x 12 steps, one workgroup ----------------
// 512 threads = 16 wave32s. Wave w owns output rows 32w..32w+31 (tm = 2w, 2w+1)
// and ALL hidden columns, so the h fragments needed at step t+1 are the rows this
// wave wrote at step t -> h lives in a single wave-private 64 KB LDS buffer, no
// per-step barrier (same-wave DS ordering suffices).
// Layer weights (wx 32 KB + wh 32 KB) are staged in LDS once per layer; an empty
// asm memory clobber inside the t-loop stops LICM from hoisting the B-fragment
// loads into (spilled) registers. h0/c0 are zeroed at layer start so the time
// loop is completely branch-free.
__global__ __launch_bounds__(512) void k_lstm_all(
        __bf16* __restrict__ seqA, __bf16* __restrict__ seqB,
        const __bf16* __restrict__ WxT_all, const __bf16* __restrict__ WhT_all,
        const float* __restrict__ b_all) {
    __shared__ __align__(64) __bf16 hsh[Mq * GHq];     // 64 KB, row-partitioned by wave
    __shared__ __align__(64) __bf16 wxs[ZNq * GHq];    // 32 KB
    __shared__ __align__(64) __bf16 whs[ZNq * GHq];    // 32 KB
    const int tid  = threadIdx.x;
    const int w    = tid >> 5;             // 0..15
    const int lane = tid & 31;

    float creg[2][4][8];

#pragma unroll 1
    for (int l = 0; l < NLq; ++l) {
        const __bf16* seqIn  = (l & 1) ? seqB : seqA;
        __bf16*       seqOut = (l & 1) ? seqA : seqB;
        const __bf16* wx   = WxT_all + (size_t)l * ZNq * GHq;
        const __bf16* wh   = WhT_all + (size_t)l * ZNq * GHq;
        const float*  bias = b_all   + (size_t)l * ZNq;

        __syncthreads();   // all waves finished with previous layer's LDS contents
        // stage this layer's weights in LDS; zero the h buffer (h0 == 0)
        for (int i = tid; i < (ZNq * GHq) / 8; i += 512) {
            ((uint4*)wxs)[i] = ((const uint4*)wx)[i];
            ((uint4*)whs)[i] = ((const uint4*)wh)[i];
        }
        uint4 z4; z4.x = z4.y = z4.z = z4.w = 0u;
        for (int i = tid; i < (Mq * GHq) / 8; i += 512) ((uint4*)hsh)[i] = z4;
#pragma unroll
        for (int bi = 0; bi < 2; ++bi)
#pragma unroll
            for (int hj = 0; hj < 4; ++hj)
#pragma unroll
                for (int v = 0; v < 8; ++v) creg[bi][hj][v] = 0.f;
        __syncthreads();

#pragma unroll 1
        for (int t = 0; t < Tq; ++t) {
            asm volatile("" ::: "memory");   // keep all loads inside the loop (no LICM+spill)
            const __bf16* xbase = seqIn  + (size_t)t * Mq * GHq;
            __bf16*       obase = seqOut + (size_t)t * Mq * GHq;

#pragma unroll
            for (int bi = 0; bi < 2; ++bi) {
                const int tm = 2 * w + bi;           // owned M block

                v16bf xa0 = ld_frag64(xbase + (size_t)tm * 16 * GHq, lane, 0);
                v16bf xa1 = ld_frag64(xbase + (size_t)tm * 16 * GHq, lane, 1);
                v16bf ha0 = ld_frag64(&hsh[tm * 16 * GHq], lane, 0);   // rows written at t-1
                v16bf ha1 = ld_frag64(&hsh[tm * 16 * GHq], lane, 1);

#pragma unroll
                for (int hj = 0; hj < 4; ++hj) {
                    v8f acc[4];
#pragma unroll
                    for (int g = 0; g < 4; ++g) {
                        const int n0 = g * GHq + hj * 16;
                        v8f a = {};
                        a = wmma_bf16(xa0, ld_frag64(wxs + (size_t)n0 * GHq, lane, 0), a);
                        a = wmma_bf16(xa1, ld_frag64(wxs + (size_t)n0 * GHq, lane, 1), a);
                        a = wmma_bf16(ha0, ld_frag64(whs + (size_t)n0 * GHq, lane, 0), a);
                        a = wmma_bf16(ha1, ld_frag64(whs + (size_t)n0 * GHq, lane, 1), a);
                        float bb = bias[n0 + (lane & 15)];
#pragma unroll
                        for (int v = 0; v < 8; ++v) a[v] += bb;
                        acc[g] = a;
                    }

                    const int h = hj * 16 + (lane & 15);
#pragma unroll
                    for (int v = 0; v < 8; ++v) {
                        const int m = tm * 16 + v + ((lane >> 4) << 3);
                        float iv = sigmoidf_(acc[0][v]);
                        float fv = sigmoidf_(acc[1][v]);
                        float gv = tanhf(acc[2][v]);
                        float ov = sigmoidf_(acc[3][v]);
                        float cn = fv * creg[bi][hj][v] + iv * gv;
                        creg[bi][hj][v] = cn;
                        __bf16 hv = (__bf16)(ov * tanhf(cn));
                        hsh[m * GHq + h] = hv;               // wave-private LDS (next step)
                        obase[(size_t)m * GHq + h] = hv;     // global mirror (next layer)
                    }
                }
            }
        }
        // make this layer's global h-sequence visible to all waves for the next layer
        __threadfence_block();
    }
}

// ---------------- 4) gF = hT @ W_out + b_out  -> cat[:, 512:1024] ----------------
__global__ void k_gf(const __bf16* __restrict__ hT, const __bf16* __restrict__ WoutT,
                     const float* __restrict__ bout, float* __restrict__ cat) {
    int gt   = blockIdx.x * blockDim.x + threadIdx.x;
    int wave = gt >> 5;                  // 0..1023
    int lane = threadIdx.x & 31;
    int tm = wave >> 5;                  // 0..31
    int tn = wave & 31;                  // 0..31
    const __bf16* a = hT + (size_t)tm * 16 * GHq;
    v16bf a0 = ld_frag64(a, lane, 0);
    v16bf a1 = ld_frag64(a, lane, 1);
    v16bf b0 = ld_frag64(WoutT + (size_t)tn * 16 * GHq, lane, 0);
    v16bf b1 = ld_frag64(WoutT + (size_t)tn * 16 * GHq, lane, 1);
    v8f acc = {};
    acc = wmma_bf16(a0, b0, acc);
    acc = wmma_bf16(a1, b1, acc);
    int n = tn * 16 + (lane & 15);
    float bb = bout[n];
#pragma unroll
    for (int v = 0; v < 8; ++v) {
        int m = tm * 16 + v + ((lane >> 4) << 3);
        cat[(size_t)m * (FCHq + OUTq) + FCHq + n] = acc[v] + bb;
    }
}

// ---------------- 5) tmp0 = x2f @ W0 + b0 ----------------
__global__ void k_fc0(const float* __restrict__ x, const float* __restrict__ W0,
                      const float* __restrict__ b0, float* __restrict__ tmp0) {
    int idx = blockIdx.x * blockDim.x + threadIdx.x;   // < 512*512
    int b = idx >> 9, j = idx & 511;
    const float* xr = x + (size_t)b * XSTR + Tq * Nq;  // x2f = x[:, 2400:2415]
    float s = b0[j];
#pragma unroll
    for (int k = 0; k < Fq - 1; ++k) s += xr[k] * W0[k * FCHq + j];
    tmp0[idx] = s;
}

// ---------------- 6) BN over batch + leaky_relu -> cat[:, 0:512] ----------------
__global__ void k_bn0(const float* __restrict__ tmp0, const float* __restrict__ g0,
                      const float* __restrict__ be0, float* __restrict__ cat) {
    int j = blockIdx.x, tid = threadIdx.x;             // 256 threads, 512 rows
    float a = tmp0[(size_t)tid * FCHq + j];
    float b = tmp0[(size_t)(tid + 256) * FCHq + j];
    __shared__ float ss[256], sq[256];
    ss[tid] = a + b; sq[tid] = a * a + b * b;
    __syncthreads();
    for (int s = 128; s > 0; s >>= 1) {
        if (tid < s) { ss[tid] += ss[tid + s]; sq[tid] += sq[tid + s]; }
        __syncthreads();
    }
    float mean = ss[0] * (1.0f / Bq);
    float var  = sq[0] * (1.0f / Bq) - mean * mean;
    float rstd = rsqrtf(var + 1e-5f);
    float g = g0[j], be = be0[j];
    float h1 = g * (a - mean) * rstd + be; h1 = h1 > 0.f ? h1 : 0.01f * h1;
    float h2 = g * (b - mean) * rstd + be; h2 = h2 > 0.f ? h2 : 0.01f * h2;
    cat[(size_t)tid * (FCHq + OUTq) + j] = h1;
    cat[(size_t)(tid + 256) * (FCHq + OUTq) + j] = h2;
}

// ---------------- 7) BN1 column stats over cat ----------------
__global__ void k_bn1_stats(const float* __restrict__ cat, float* __restrict__ mean1,
                            float* __restrict__ rstd1) {
    int j = blockIdx.x, tid = threadIdx.x;             // 1024 blocks
    float a = cat[(size_t)tid * (FCHq + OUTq) + j];
    float b = cat[(size_t)(tid + 256) * (FCHq + OUTq) + j];
    __shared__ float ss[256], sq[256];
    ss[tid] = a + b; sq[tid] = a * a + b * b;
    __syncthreads();
    for (int s = 128; s > 0; s >>= 1) {
        if (tid < s) { ss[tid] += ss[tid + s]; sq[tid] += sq[tid + s]; }
        __syncthreads();
    }
    if (tid == 0) {
        float mean = ss[0] * (1.0f / Bq);
        float var  = sq[0] * (1.0f / Bq) - mean * mean;
        mean1[j] = mean;
        rstd1[j] = rsqrtf(var + 1e-5f);
    }
}

// ---------------- 8) out[b] = sigmoid( bn1(cat[b]) @ W1 + b1 ) ----------------
__global__ void k_final(const float* __restrict__ cat, const float* __restrict__ mean1,
                        const float* __restrict__ rstd1, const float* __restrict__ g1,
                        const float* __restrict__ be1, const float* __restrict__ W1,
                        const float* __restrict__ b1, float* __restrict__ out) {
    int b = blockIdx.x, tid = threadIdx.x;             // 256 threads
    float s = 0.f;
    for (int j = tid; j < FCHq + OUTq; j += 256) {
        float v = (cat[(size_t)b * (FCHq + OUTq) + j] - mean1[j]) * rstd1[j] * g1[j] + be1[j];
        s += v * W1[j];
    }
    __shared__ float ss[256];
    ss[tid] = s;
    __syncthreads();
    for (int st = 128; st > 0; st >>= 1) {
        if (tid < st) ss[tid] += ss[tid + st];
        __syncthreads();
    }
    if (tid == 0) out[b] = 1.0f / (1.0f + expf(-(ss[0] + b1[0])));
}

// ---------------- host launcher ----------------
extern "C" void kernel_launch(void* const* d_in, const int* in_sizes, int n_in,
                              void* d_out, int out_size, void* d_ws, size_t ws_size,
                              hipStream_t stream) {
    const float* x      = (const float*)d_in[0];
    const float* cheb   = (const float*)d_in[1];
    const float* W_g    = (const float*)d_in[2];
    const float* b_g    = (const float*)d_in[3];
    const float* Wx     = (const float*)d_in[4];
    const float* Wh     = (const float*)d_in[5];
    const float* b_lstm = (const float*)d_in[6];
    const float* W_out  = (const float*)d_in[7];
    const float* b_out  = (const float*)d_in[8];
    const float* W0     = (const float*)d_in[9];
    const float* b0     = (const float*)d_in[10];
    const float* g0     = (const float*)d_in[11];
    const float* be0    = (const float*)d_in[12];
    const float* g1     = (const float*)d_in[13];
    const float* be1    = (const float*)d_in[14];
    const float* W1     = (const float*)d_in[15];
    const float* b1     = (const float*)d_in[16];
    float* out = (float*)d_out;

    char* ws = (char*)d_ws;
    size_t off = 0;
    auto alloc = [&](size_t bytes) { char* p = ws + off; off = (off + bytes + 255) & ~(size_t)255; return p; };
    __bf16* WxT   = (__bf16*)alloc((size_t)NLq * ZNq * GHq * 2);      // 98304 B
    __bf16* WhT   = (__bf16*)alloc((size_t)NLq * ZNq * GHq * 2);
    __bf16* WoutT = (__bf16*)alloc((size_t)OUTq * GHq * 2);           // 65536 B
    __bf16* seqA  = (__bf16*)alloc((size_t)Tq * Mq * GHq * 2);        // 786432 B
    __bf16* seqB  = (__bf16*)alloc((size_t)Tq * Mq * GHq * 2);
    float*  tmp0  = (float*) alloc((size_t)Bq * FCHq * 4);            // 1 MB
    float*  cat   = (float*) alloc((size_t)Bq * (FCHq + OUTq) * 4);   // 2 MB
    float*  mean1 = (float*) alloc(1024 * 4);
    float*  rstd1 = (float*) alloc(1024 * 4);
    (void)ws_size; (void)n_in; (void)in_sizes; (void)out_size;

    // 1) weight prep (131072 elements)
    k_prep<<<512, 256, 0, stream>>>(Wx, Wh, W_out, WxT, WhT, WoutT);

    // 2) gathered graph conv -> seqA
    k_gconv<<<dim3(Tq, Bq), 64, 0, stream>>>(x, cheb, W_g, b_g, seqA);

    // 3) persistent LSTM (3 layers x 12 steps in one launch)
    k_lstm_all<<<1, 512, 0, stream>>>(seqA, seqB, WxT, WhT, b_lstm);
    const __bf16* hT = seqB + (size_t)(Tq - 1) * Mq * GHq;   // layer 2 writes seqB

    // 4) gF -> cat[:, 512:1024]
    k_gf<<<256, 128, 0, stream>>>(hT, WoutT, b_out, cat);

    // 5-6) MLP branch -> cat[:, 0:512]
    k_fc0<<<1024, 256, 0, stream>>>(x, W0, b0, tmp0);
    k_bn0<<<512, 256, 0, stream>>>(tmp0, g0, be0, cat);

    // 7-8) BN1 + final sigmoid dot
    k_bn1_stats<<<1024, 256, 0, stream>>>(cat, mean1, rstd1);
    k_final<<<512, 256, 0, stream>>>(cat, mean1, rstd1, g1, be1, W1, b1, out);
}
